// HSMSSDEnhancedAttentionLayer_79852031967736
// MI455X (gfx1250) — matrix operand order repaired and checked by reading, not compiled, not run on hardware
//
#include <hip/hip_runtime.h>
#include <hip/hip_bf16.h>

// ---------------------------------------------------------------------------
// HSMSSD Enhanced Attention layer for MI455X (gfx1250, wave32, WMMA).
//
//   Wh = h @ W                            (WMMA f16 GEMM, f32 accum)
//   P[i,j] = softmax_j(mask(leaky(s_src[i]+s_dst[j])))  f16, fits 192MB L2
//   h_attn = P @ Wh                       (dominant 34 GFLOP WMMA GEMM)
//   + feature enhancer / SSM branch / gated fusion via skinny WMMA GEMMs.
//
// GEMM design: B operands stored transposed (BT[N][K]) so B fragments are
// contiguous; BT tiles staged to LDS with gfx1250 async-to-LDS (ASYNCcnt,
// inline asm: global_load_async_to_lds_b128 + s_wait_asynccnt), shared by
// 4 waves; A fragments are contiguous b128 loads.
// ---------------------------------------------------------------------------

typedef __attribute__((ext_vector_type(16))) _Float16 v16h;
typedef __attribute__((ext_vector_type(8)))  _Float16 v8h;
typedef __attribute__((ext_vector_type(8)))  float    v8f;

#define NN   8192
#define FF   256
#define SS   128
#define DIN  512

// ---- gfx1250 async global->LDS staging (inline asm, ASYNCcnt-tracked) -----
// Generic LDS pointers are {aperture_hi, lds_offset} (ISA 10.2): low 32 bits
// give the LDS byte address the instruction's VDST VGPR needs.
__device__ __forceinline__ void async_copy16(const void* g, void* lds)
{
    const unsigned lds_off = (unsigned)(unsigned long long)lds;
    asm volatile("global_load_async_to_lds_b128 %0, %1, off"
                 :: "v"(lds_off), "v"(g)
                 : "memory");
}

__device__ __forceinline__ void wait_async()
{
    asm volatile("s_wait_asynccnt 0x0" ::: "memory");
}

// ---------------------------------------------------------------------------
// WMMA GEMM:  C[M,Nc] (f32,+bias) = A[M,K] (f16 row-major) @ B (given as
// BT[Nc,K], f16).  Block = 128 threads = 4 waves -> 64x32 C tile.
// Wave w: rows m0+16w..+15, cols n0..n0+31 (two 16x16 accumulators).
// BT tile (32 cols x 32 k = 2 KB) staged in LDS per k-step (async), 4x reuse.
// M % 64 == 0, Nc % 32 == 0, K % 32 == 0 (true for all calls here).
// ---------------------------------------------------------------------------
__global__ __launch_bounds__(128)
void wmma_gemm_kernel(const _Float16* __restrict__ A,
                      const _Float16* __restrict__ BT,
                      const float*    __restrict__ bias,
                      float*          __restrict__ C,
                      int M, int K, int Nc)
{
    __shared__ _Float16 sBT[32 * 32];   // [n_local][k_local], 2 KB

    const int tid  = threadIdx.x;
    const int wave = tid >> 5;
    const int lane = tid & 31;
    const int half = lane >> 4;          // 0: lanes 0-15, 1: lanes 16-31
    const int l15  = lane & 15;

    const int n0 = blockIdx.x * 32;
    const int m0 = blockIdx.y * 64 + wave * 16;
    const int m  = m0 + l15;

    // staging: thread t copies 16B: n_local = t/4, k8 = (t%4)*8
    const int sn = tid >> 2;
    const int sk = (tid & 3) * 8;
    const _Float16* gB = BT + (size_t)(n0 + sn) * K + sk;
    _Float16*       lB = sBT + sn * 32 + sk;

    const _Float16* Arow = A + (size_t)m * K;

    v8f acc0 = {}, acc1 = {};

    for (int k0 = 0; k0 < K; k0 += 32) {
        __syncthreads();                     // prior-iteration LDS reads done
        async_copy16(gB + k0, lB);           // stage BT tile (ASYNCcnt)

        // A fragment (16x32 f16): lane holds row m; element e -> K = half*16 + e
        // (contiguous, matching B): two 8-half runs at k0+half*16 and
        // k0+half*16+8.
        const v8h alo = *(const v8h*)(Arow + k0 + half * 16);
        const v8h ahi = *(const v8h*)(Arow + k0 + half * 16 + 8);
        const v16h a = __builtin_shufflevector(alo, ahi,
            0, 1, 2, 3, 4, 5, 6, 7, 8, 9, 10, 11, 12, 13, 14, 15);

        if (k0 + 32 < K) __builtin_prefetch(Arow + k0 + 32, 0, 1);

        wait_async();
        __syncthreads();                     // BT tile visible to all waves

        // B fragments: lane col n; element e -> K = half*16 + e (contiguous)
        const v16h b0 = *(const v16h*)(sBT + l15 * 32 + half * 16);
        const v16h b1 = *(const v16h*)(sBT + (16 + l15) * 32 + half * 16);

        acc0 = __builtin_amdgcn_wmma_f32_16x16x32_f16(
            false, a, false, b0, (short)0, acc0, false, false);
        acc1 = __builtin_amdgcn_wmma_f32_16x16x32_f16(
            false, a, false, b1, (short)0, acc1, false, false);
    }

    const float bn0 = bias ? bias[n0 + l15] : 0.0f;
    const float bn1 = bias ? bias[n0 + 16 + l15] : 0.0f;
#pragma unroll
    for (int v = 0; v < 8; ++v) {
        const int row = m0 + v + 8 * half;
        C[(size_t)row * Nc + n0 + l15]      = acc0[v] + bn0;
        C[(size_t)row * Nc + n0 + 16 + l15] = acc1[v] + bn1;
    }
}

// ---------------------------------------------------------------------------
// f32 -> f16 (A-side operands)
// ---------------------------------------------------------------------------
__global__ void f32_to_f16_kernel(const float* __restrict__ x,
                                  _Float16* __restrict__ y, int n)
{
    int i = blockIdx.x * blockDim.x + threadIdx.x;
    if (i < n) y[i] = (_Float16)x[i];
}

// f32 [R,Cc] -> f16 transposed [Cc,R]  (B-side operands)
__global__ void f32_to_f16_T_kernel(const float* __restrict__ x,
                                    _Float16* __restrict__ y, int R, int Cc)
{
    int idx = blockIdx.x * blockDim.x + threadIdx.x;
    if (idx >= R * Cc) return;
    const int r = idx / Cc, c = idx % Cc;
    y[(size_t)c * R + r] = (_Float16)x[idx];
}

// ---------------------------------------------------------------------------
// s_src[i] = dot(Wh[i,:], a_src);  s_dst[i] = dot(Wh[i,:], a_dst)
// ---------------------------------------------------------------------------
__global__ __launch_bounds__(256)
void rowvec_kernel(const float* __restrict__ Wh,
                   const float* __restrict__ a_src,
                   const float* __restrict__ a_dst,
                   float* __restrict__ ssrc, float* __restrict__ sdst)
{
    __shared__ float r1[256];
    __shared__ float r2[256];
    const int i = blockIdx.x, t = threadIdx.x;
    const float w = Wh[(size_t)i * FF + t];
    r1[t] = w * a_src[t];
    r2[t] = w * a_dst[t];
    __syncthreads();
    for (int k = 128; k > 0; k >>= 1) {
        if (t < k) { r1[t] += r1[t + k]; r2[t] += r2[t + k]; }
        __syncthreads();
    }
    if (t == 0) { ssrc[i] = r1[0]; sdst[i] = r2[0]; }
}

// ---------------------------------------------------------------------------
// P[i,:] = softmax(mask(leaky(s_src[i]+s_dst))) as f16; row cached in LDS.
// ---------------------------------------------------------------------------
__global__ __launch_bounds__(256)
void attn_p_kernel(const float* __restrict__ ssrc,
                   const float* __restrict__ sdst,
                   const int*   __restrict__ adj,
                   _Float16*    __restrict__ P)
{
    __shared__ float se[NN];
    __shared__ float red[256];
    const int i = blockIdx.x, t = threadIdx.x;
    const float si = ssrc[i];
    const int* arow = adj + (size_t)i * NN;

    float mx = -3.4e38f;
    for (int j = t; j < NN; j += 256) {
        float e = si + sdst[j];
        e = (e > 0.0f) ? e : 0.2f * e;
        if (arow[j] <= 0) e = -9.0e15f;
        se[j] = e;
        mx = fmaxf(mx, e);
    }
    red[t] = mx;
    __syncthreads();
    for (int k = 128; k > 0; k >>= 1) {
        if (t < k) red[t] = fmaxf(red[t], red[t + k]);
        __syncthreads();
    }
    const float rm = red[0];
    __syncthreads();

    float sum = 0.0f;
    for (int j = t; j < NN; j += 256) {
        float p = __expf(se[j] - rm);
        se[j] = p;
        sum += p;
    }
    red[t] = sum;
    __syncthreads();
    for (int k = 128; k > 0; k >>= 1) {
        if (t < k) red[t] += red[t + k];
        __syncthreads();
    }
    const float inv = 1.0f / red[0];

    _Float16* prow = P + (size_t)i * NN;
    for (int j = t; j < NN; j += 256) {
        prow[j] = (_Float16)(se[j] * inv);
    }
}

// ---------------------------------------------------------------------------
// LayerNorm + SiLU -> f16, per row.
// ---------------------------------------------------------------------------
__global__ __launch_bounds__(256)
void ln_silu_f16_kernel(const float* __restrict__ X,
                        const float* __restrict__ g,
                        const float* __restrict__ b,
                        _Float16* __restrict__ Y, int D)
{
    __shared__ float red[256];
    const int i = blockIdx.x, t = threadIdx.x;
    const float* x = X + (size_t)i * D;

    float s = 0.0f;
    for (int c = t; c < D; c += 256) s += x[c];
    red[t] = s;
    __syncthreads();
    for (int k = 128; k > 0; k >>= 1) { if (t < k) red[t] += red[t + k]; __syncthreads(); }
    const float mean = red[0] / (float)D;
    __syncthreads();

    float v = 0.0f;
    for (int c = t; c < D; c += 256) { float d = x[c] - mean; v += d * d; }
    red[t] = v;
    __syncthreads();
    for (int k = 128; k > 0; k >>= 1) { if (t < k) red[t] += red[t + k]; __syncthreads(); }
    const float rstd = rsqrtf(red[0] / (float)D + 1e-5f);

    _Float16* y = Y + (size_t)i * D;
    for (int c = t; c < D; c += 256) {
        float u = (x[c] - mean) * rstd * g[c] + b[c];
        y[c] = (_Float16)(u / (1.0f + __expf(-u)));   // silu
    }
}

// ---------------------------------------------------------------------------
// SSM mid: Asm = softmax(dt + A); AsmB = Asm*B (f16); cvec = C@cp_w + cp_b.
// ---------------------------------------------------------------------------
__global__ __launch_bounds__(128)
void ssm_mid_kernel(const float* __restrict__ BCdt,
                    const float* __restrict__ Avec,
                    const float* __restrict__ cp_w,
                    const float* __restrict__ cp_b,
                    _Float16* __restrict__ AsmB,
                    float* __restrict__ cvec)
{
    __shared__ float red[128];
    const int i = blockIdx.x, t = threadIdx.x;
    const float* row = BCdt + (size_t)i * (3 * SS);
    const float Bv = row[t];
    const float Cv = row[SS + t];
    const float d  = row[2 * SS + t] + Avec[t];

    red[t] = d;
    __syncthreads();
    for (int k = 64; k > 0; k >>= 1) { if (t < k) red[t] = fmaxf(red[t], red[t + k]); __syncthreads(); }
    const float mx = red[0];
    __syncthreads();

    const float ex = __expf(d - mx);
    red[t] = ex;
    __syncthreads();
    for (int k = 64; k > 0; k >>= 1) { if (t < k) red[t] += red[t + k]; __syncthreads(); }
    const float inv = 1.0f / red[0];
    __syncthreads();

    AsmB[(size_t)i * SS + t] = (_Float16)(ex * inv * Bv);

    red[t] = Cv * cp_w[t];
    __syncthreads();
    for (int k = 64; k > 0; k >>= 1) { if (t < k) red[t] += red[t + k]; __syncthreads(); }
    if (t == 0) cvec[i] = red[0] + cp_b[0];
}

__global__ void mul_f16_kernel(const float* __restrict__ a,
                               const float* __restrict__ b,
                               _Float16* __restrict__ y, int n)
{
    int i = blockIdx.x * blockDim.x + threadIdx.x;
    if (i < n) y[i] = (_Float16)(a[i] * b[i]);
}

// GLU: u = h_part*silu(z) + h_part*D  (hz[N,1024] -> u16[N,512])
__global__ void glu_f16_kernel(const float* __restrict__ HZ,
                               const float* __restrict__ Dp,
                               _Float16* __restrict__ U, int total)
{
    int idx = blockIdx.x * blockDim.x + threadIdx.x;
    if (idx >= total) return;
    const int row = idx >> 9;
    const int c   = idx & 511;
    const float* r = HZ + (size_t)row * (2 * DIN);
    const float hp = r[c];
    const float z  = r[DIN + c];
    const float sz = z / (1.0f + __expf(-z));
    U[idx] = (_Float16)(hp * sz + hp * Dp[0]);
}

// hs_s = hs*(1+cvec[row]); comb16 = [h_attn | hs_s]
__global__ void scale_concat_kernel(const float* __restrict__ hs,
                                    const float* __restrict__ cvec,
                                    const float* __restrict__ h_attn,
                                    float* __restrict__ hs_s,
                                    _Float16* __restrict__ comb, int total)
{
    int idx = blockIdx.x * blockDim.x + threadIdx.x;
    if (idx >= total) return;
    const int row = idx >> 8;
    const int c   = idx & 255;
    const float v = hs[idx] * (1.0f + cvec[row]);
    hs_s[idx] = v;
    comb[(size_t)row * (2 * FF) + c]      = (_Float16)h_attn[idx];
    comb[(size_t)row * (2 * FF) + FF + c] = (_Float16)v;
}

// g = sigmoid(gpre); fused = g*h_attn + (1-g)*hs_s + h_enh; final LayerNorm.
__global__ __launch_bounds__(256)
void fuse_ln_kernel(const float* __restrict__ gpre,
                    const float* __restrict__ h_attn,
                    const float* __restrict__ hs_s,
                    const float* __restrict__ h_enh,
                    const float* __restrict__ lng,
                    const float* __restrict__ lnb,
                    float* __restrict__ out)
{
    __shared__ float red[256];
    const int i = blockIdx.x, t = threadIdx.x;
    const size_t o = (size_t)i * FF + t;
    const float gg = 1.0f / (1.0f + __expf(-gpre[o]));
    const float f  = gg * h_attn[o] + (1.0f - gg) * hs_s[o] + h_enh[o];

    red[t] = f;
    __syncthreads();
    for (int k = 128; k > 0; k >>= 1) { if (t < k) red[t] += red[t + k]; __syncthreads(); }
    const float mean = red[0] / (float)FF;
    __syncthreads();

    const float d = f - mean;
    red[t] = d * d;
    __syncthreads();
    for (int k = 128; k > 0; k >>= 1) { if (t < k) red[t] += red[t + k]; __syncthreads(); }
    const float rstd = rsqrtf(red[0] / (float)FF + 1e-5f);

    out[o] = d * rstd * lng[t] + lnb[t];
}

// ---------------------------------------------------------------------------
// Host-side orchestration
// ---------------------------------------------------------------------------
static inline size_t al256(size_t x) { return (x + 255) & ~(size_t)255; }

extern "C" void kernel_launch(void* const* d_in, const int* in_sizes, int n_in,
                              void* d_out, int out_size, void* d_ws, size_t ws_size,
                              hipStream_t stream)
{
    (void)in_sizes; (void)n_in; (void)out_size; (void)ws_size;

    const float* h      = (const float*)d_in[0];
    const int*   adj    = (const int*)  d_in[1];
    const float* W      = (const float*)d_in[2];
    const float* a_src  = (const float*)d_in[3];
    const float* a_dst  = (const float*)d_in[4];
    const float* W_bcdt = (const float*)d_in[5];
    const float* b_bcdt = (const float*)d_in[6];
    const float* abp_w  = (const float*)d_in[7];
    const float* abp_b  = (const float*)d_in[8];
    const float* cp_w   = (const float*)d_in[9];
    const float* cp_b   = (const float*)d_in[10];
    const float* W_hz   = (const float*)d_in[11];
    const float* b_hz   = (const float*)d_in[12];
    const float* out_w  = (const float*)d_in[13];
    const float* out_b  = (const float*)d_in[14];
    const float* fe1_w  = (const float*)d_in[15];
    const float* fe1_b  = (const float*)d_in[16];
    const float* fe_ln_g= (const float*)d_in[17];
    const float* fe_ln_b= (const float*)d_in[18];
    const float* fe2_w  = (const float*)d_in[19];
    const float* fe2_b  = (const float*)d_in[20];
    const float* Avec   = (const float*)d_in[21];
    const float* Dsc    = (const float*)d_in[22];
    const float* g1_w   = (const float*)d_in[23];
    const float* g1_b   = (const float*)d_in[24];
    const float* g_ln_g = (const float*)d_in[25];
    const float* g_ln_b = (const float*)d_in[26];
    const float* g2_w   = (const float*)d_in[27];
    const float* g2_b   = (const float*)d_in[28];
    const float* ln_g   = (const float*)d_in[29];
    const float* ln_b   = (const float*)d_in[30];

    float* outp = (float*)d_out;

    // ---- workspace carve-up ----
    char* ws = (char*)d_ws;
    size_t off = 0;
    auto take = [&](size_t bytes) { void* p = ws + off; off += al256(bytes); return p; };

    _Float16* P16    = (_Float16*)take((size_t)NN * NN * 2);     // 128 MB
    _Float16* h16    = (_Float16*)take((size_t)NN * FF * 2);
    float*    Wh     = (float*)   take((size_t)NN * FF * 4);
    _Float16* Wh16   = (_Float16*)take((size_t)NN * FF * 2);    // row-major (A side)
    _Float16* WhT16  = (_Float16*)take((size_t)NN * FF * 2);    // transposed (B side)
    float*    ssrc   = (float*)   take((size_t)NN * 4);
    float*    sdst   = (float*)   take((size_t)NN * 4);
    float*    cvec   = (float*)   take((size_t)NN * 4);
    float*    h_attn = (float*)   take((size_t)NN * FF * 4);
    float*    h_enh  = (float*)   take((size_t)NN * FF * 4);
    float*    hs     = (float*)   take((size_t)NN * FF * 4);
    float*    hs_s   = (float*)   take((size_t)NN * FF * 4);
    float*    gpre   = (float*)   take((size_t)NN * FF * 4);
    float*    ab     = (float*)   take((size_t)NN * FF * 4);
    float*    bigf   = (float*)   take((size_t)NN * 1024 * 4);  // t1/BCdt/hz/gmid
    _Float16* b16    = (_Float16*)take((size_t)NN * DIN * 2);   // rotating f16 buf
    // transposed f16 weight copies (BT[N][K])
    _Float16* W16T    = (_Float16*)take((size_t)FF * FF * 2);
    _Float16* fe1T    = (_Float16*)take((size_t)FF * DIN * 2);
    _Float16* fe2T    = (_Float16*)take((size_t)DIN * FF * 2);
    _Float16* bcdtT   = (_Float16*)take((size_t)FF * 3 * SS * 2);
    _Float16* abpT    = (_Float16*)take((size_t)SS * FF * 2);
    _Float16* hzT     = (_Float16*)take((size_t)FF * 2 * DIN * 2);
    _Float16* outwT   = (_Float16*)take((size_t)DIN * FF * 2);
    _Float16* g1T     = (_Float16*)take((size_t)(2 * FF) * (2 * FF) * 2);
    _Float16* g2T     = (_Float16*)take((size_t)(2 * FF) * FF * 2);

    auto cvt = [&](const float* src, _Float16* dst, int n) {
        f32_to_f16_kernel<<<(n + 255) / 256, 256, 0, stream>>>(src, dst, n);
    };
    auto cvtT = [&](const float* src, _Float16* dst, int R, int Cc) {
        f32_to_f16_T_kernel<<<(R * Cc + 255) / 256, 256, 0, stream>>>(src, dst, R, Cc);
    };
    auto gemm = [&](const _Float16* A, const _Float16* BT, const float* bias,
                    float* C, int M, int K, int Nc) {
        dim3 grid(Nc / 32, M / 64);
        wmma_gemm_kernel<<<grid, 128, 0, stream>>>(A, BT, bias, C, M, K, Nc);
    };

    // 0) f16 copies: A-side row-major, B-side transposed
    cvt (h, h16, NN * FF);
    cvtT(W,      W16T,  FF,  FF);
    cvtT(fe1_w,  fe1T,  FF,  DIN);
    cvtT(fe2_w,  fe2T,  DIN, FF);
    cvtT(W_bcdt, bcdtT, FF,  3 * SS);
    cvtT(abp_w,  abpT,  SS,  FF);
    cvtT(W_hz,   hzT,   FF,  2 * DIN);
    cvtT(out_w,  outwT, DIN, FF);
    cvtT(g1_w,   g1T,   2 * FF, 2 * FF);
    cvtT(g2_w,   g2T,   2 * FF, FF);

    // 1) Wh = h @ W
    gemm(h16, W16T, nullptr, Wh, NN, FF, FF);
    cvt (Wh, Wh16, NN * FF);
    cvtT(Wh, WhT16, NN, FF);

    // 2) s_src / s_dst
    rowvec_kernel<<<NN, 256, 0, stream>>>(Wh, a_src, a_dst, ssrc, sdst);

    // 3) attention probabilities P (f16, row-wise masked softmax)
    attn_p_kernel<<<NN, 256, 0, stream>>>(ssrc, sdst, adj, P16);

    // 4) h_attn = P @ Wh          (dominant GEMM: 8192 x 8192 x 256)
    gemm(P16, WhT16, nullptr, h_attn, NN, NN, FF);

    // 5) feature enhancer: silu(LN(Wh@fe1+b)) @ fe2 + b
    gemm(Wh16, fe1T, fe1_b, bigf, NN, FF, DIN);
    ln_silu_f16_kernel<<<NN, 256, 0, stream>>>(bigf, fe_ln_g, fe_ln_b, b16, DIN);
    gemm(b16, fe2T, fe2_b, h_enh, NN, DIN, FF);

    // 6) SSM branch
    gemm(Wh16, bcdtT, b_bcdt, bigf, NN, FF, 3 * SS);
    ssm_mid_kernel<<<NN, 128, 0, stream>>>(bigf, Avec, cp_w, cp_b, b16, cvec);
    gemm(b16, abpT, abp_b, ab, NN, SS, FF);                     // ab = (Asm*B)@abp_w
    mul_f16_kernel<<<(NN * FF + 255) / 256, 256, 0, stream>>>(Wh, ab, b16, NN * FF);
    gemm(b16, hzT, b_hz, bigf, NN, FF, 2 * DIN);                // hz
    glu_f16_kernel<<<(NN * DIN + 255) / 256, 256, 0, stream>>>(bigf, Dsc, b16, NN * DIN);
    gemm(b16, outwT, out_b, hs, NN, DIN, FF);                   // hs (pre c_proj scale)

    // 7) gated fusion
    scale_concat_kernel<<<(NN * FF + 255) / 256, 256, 0, stream>>>(
        hs, cvec, h_attn, hs_s, b16, NN * FF);                  // b16 = comb f16
    gemm(b16, g1T, g1_b, bigf, NN, 2 * FF, 2 * FF);             // gmid
    ln_silu_f16_kernel<<<NN, 256, 0, stream>>>(bigf, g_ln_g, g_ln_b, b16, 2 * FF);
    gemm(b16, g2T, g2_b, gpre, NN, 2 * FF, FF);                 // pre-sigmoid g

    // 8) fuse + final LN -> output
    fuse_ln_kernel<<<NN, 256, 0, stream>>>(gpre, h_attn, hs_s, h_enh, ln_g, ln_b, outp);
}
